// GATRegression_56650618634887
// MI455X (gfx1250) — matrix operand order repaired and checked by reading, not compiled, not run on hardware
//
#include <hip/hip_runtime.h>
#include <cstddef>

// ---------------------------------------------------------------------------
// GAT 2-layer forward for MI455X (gfx1250, wave32).
//   Layer1: h1 = x@W1 (bf16 WMMA, f32 acc)  -> attention softmax aggregate
//   Layer2: z = relu(agg/den + b1) @ W2     -> scalar attention -> sigmoid
// ---------------------------------------------------------------------------

#define NN     50000
#define EE     800000
#define ETOT   (EE + NN)      // self loops appended
#define CIN    128
#define HID    128
#define NHEAD  2
#define HOUT   (NHEAD * HID)  // 256
#define NEG_SLOPE 0.2f
#define ORD_NEG_INF 0x007FFFFFu   // f2ord(-inf)

typedef __attribute__((ext_vector_type(16))) __bf16 v16bf;
typedef __attribute__((ext_vector_type(8)))  __bf16 v8bf;
typedef __attribute__((ext_vector_type(8)))  float  v8f;

__device__ __forceinline__ unsigned f2ord(float f) {
    unsigned u = __float_as_uint(f);
    return (u & 0x80000000u) ? ~u : (u | 0x80000000u);
}
__device__ __forceinline__ float ord2f(unsigned u) {
    return __uint_as_float((u & 0x80000000u) ? (u ^ 0x80000000u) : ~u);
}
__device__ __forceinline__ float leaky(float x) {
    return x > 0.0f ? x : NEG_SLOPE * x;
}

// ---- 0) convert x -> bf16 --------------------------------------------------
__global__ void k_convert_x(const float* __restrict__ x, __bf16* __restrict__ xb) {
    size_t i = (size_t)blockIdx.x * blockDim.x + threadIdx.x;
    if (i < (size_t)NN * CIN) xb[i] = (__bf16)x[i];
}

// ---- 1) pack W1 into per-lane WMMA B fragments -----------------------------
// layout: [kt(4)][nt(16)][lane(32)][i(16)] ; B 32x16 (KxN) 16-bit fragment:
//   n = lane%16 ; K = kt*32 + (lane<16 ? 0 : 8) + (i&7) + (i<8 ? 0 : 16)
__global__ void k_pack_w1(const float* __restrict__ W1, __bf16* __restrict__ w1f) {
    int idx = blockIdx.x * blockDim.x + threadIdx.x;
    if (idx >= 4 * 16 * 32 * 16) return;
    int i    = idx & 15;
    int lane = (idx >> 4) & 31;
    int nt   = (idx >> 9) & 15;
    int kt   = idx >> 13;
    int n    = nt * 16 + (lane & 15);
    int k    = kt * 32 + ((lane >> 4) ? 8 : 0) + (i & 7) + ((i >> 3) ? 16 : 0);
    w1f[idx] = (__bf16)W1[(size_t)k * HOUT + n];
}

// ---- 2) reset accumulators / ordered-max buffers (every call) --------------
__global__ void k_init(float* __restrict__ acc1, float* __restrict__ den1,
                       unsigned* __restrict__ amax1, float* __restrict__ den2,
                       float* __restrict__ num2, unsigned* __restrict__ amax2) {
    size_t i = (size_t)blockIdx.x * blockDim.x + threadIdx.x;
    if (i < (size_t)NN * HOUT) acc1[i] = 0.0f;
    if (i < (size_t)NN * NHEAD) { den1[i] = 0.0f; amax1[i] = ORD_NEG_INF; }
    if (i < (size_t)NN) { den2[i] = 0.0f; num2[i] = 0.0f; amax2[i] = ORD_NEG_INF; }
}

// ---- 3) h1 = xb @ W1  via v_wmma_f32_16x16x32_bf16 -------------------------
// One 16x16 output tile per wave; K=128 in 4 chunks of 32.
// A 16x32 bf16 fragment per ISA layout: lane<16 -> M=lane, K = {kb..kb+7, kb+16..kb+23}
//                                       lane>=16 -> M=lane-16, K shifted by +8.
__global__ __launch_bounds__(256) void k_gemm1(const __bf16* __restrict__ xb,
                                               const __bf16* __restrict__ w1f,
                                               float* __restrict__ h1) {
    const int wave = threadIdx.x >> 5;
    const int lane = threadIdx.x & 31;
    const int tile = blockIdx.x * 8 + wave;     // 50000 tiles total
    const int nt = tile & 15;
    const int mt = tile >> 4;                   // 0..3124, N=50000 divides by 16
    const int m0 = mt * 16;
    const int hi = lane >> 4;                   // lane-half select
    const int r  = lane & 15;

    v8f acc = {};
    const __bf16* arow  = xb + (size_t)(m0 + r) * CIN;
    const __bf16* bbase = w1f + ((size_t)nt * 32 + lane) * 16;

#pragma unroll
    for (int kt = 0; kt < 4; ++kt) {
        const int kb = kt * 32;
        union { v16bf v; v8bf h[2]; } A;
        A.h[0] = *(const v8bf*)(arow + kb + hi * 8);
        A.h[1] = *(const v8bf*)(arow + kb + 16 + hi * 8);
        v16bf B = *(const v16bf*)(bbase + (size_t)kt * (16 * 32 * 16));
        acc = __builtin_amdgcn_wmma_f32_16x16x32_bf16(
                  false, A.v, false, B, (short)0, acc, false, false);
    }
    // C layout: VGPR v -> M = v + hi*8, N = lane%16
    float* crow = h1 + (size_t)(m0 + hi * 8) * HOUT + nt * 16 + r;
#pragma unroll
    for (int v = 0; v < 8; ++v) crow[(size_t)v * HOUT] = acc[v];
}

// ---- 4) per-node attention dots: a_src1/a_dst1 [N,2] -----------------------
__global__ __launch_bounds__(256) void k_attn_dots(const float* __restrict__ h1,
                                                   const float* __restrict__ att_s,
                                                   const float* __restrict__ att_d,
                                                   float* __restrict__ as1,
                                                   float* __restrict__ ad1) {
    const int n = blockIdx.x;
    const int t = threadIdx.x;                 // t = head*128 + c
    __shared__ float ss[256], sd[256];
    float v = h1[(size_t)n * HOUT + t];
    ss[t] = v * att_s[t];
    sd[t] = v * att_d[t];
    __syncthreads();
    for (int off = 64; off > 0; off >>= 1) {   // reduce each 128-half
        if ((t & 127) < off) { ss[t] += ss[t + off]; sd[t] += sd[t + off]; }
        __syncthreads();
    }
    if ((t & 127) == 0) {
        int head = t >> 7;
        as1[n * NHEAD + head] = ss[t];
        ad1[n * NHEAD + head] = sd[t];
    }
}

// ---- 5) layer1 edge logits + segment max -----------------------------------
__global__ void k_edge_logits1(const int* __restrict__ ei,
                               const float* __restrict__ as1,
                               const float* __restrict__ ad1,
                               float* __restrict__ alpha1,
                               unsigned* __restrict__ amax1) {
    size_t idx = (size_t)blockIdx.x * blockDim.x + threadIdx.x;
    if (idx >= (size_t)ETOT * NHEAD) return;
    int e = (int)(idx >> 1), h = (int)(idx & 1);
    int s, d;
    if (e < EE) { s = ei[e]; d = ei[EE + e]; } else { s = d = e - EE; }
    float a = leaky(as1[s * NHEAD + h] + ad1[d * NHEAD + h]);
    alpha1[idx] = a;
    atomicMax(&amax1[d * NHEAD + h], f2ord(a));
}

// ---- 6) layer1 heavy pass: ex, denom, weighted scatter-add -----------------
__global__ __launch_bounds__(256) void k_edge_agg1(const int* __restrict__ ei,
                                                   const float* __restrict__ alpha1,
                                                   const unsigned* __restrict__ amax1,
                                                   const float* __restrict__ h1,
                                                   float* __restrict__ den1,
                                                   float* __restrict__ acc1) {
    const int e = blockIdx.x;
    const int t = threadIdx.x;
    int s, d;
    if (e < EE) { s = ei[e]; d = ei[EE + e]; } else { s = d = e - EE; }
    __shared__ float ex[NHEAD];
    if (t < NHEAD) {
        float a = alpha1[(size_t)e * NHEAD + t];
        float m = ord2f(amax1[d * NHEAD + t]);
        float v = __expf(a - m);
        ex[t] = v;
        atomicAdd(&den1[d * NHEAD + t], v);
    }
    __syncthreads();
    float w = ex[t >> 7];
    atomicAdd(&acc1[(size_t)d * HOUT + t], w * h1[(size_t)s * HOUT + t]);
}

// ---- 7) finalize layer1 + layer2 projection (fused) ------------------------
// h2 = relu(acc/den + b1); z = h2 . W2 ; a_src2/a_dst2 scalars
__global__ __launch_bounds__(256) void k_finalize1(const float* __restrict__ acc1,
                                                   const float* __restrict__ den1,
                                                   const float* __restrict__ b1,
                                                   const float* __restrict__ W2,
                                                   const float* __restrict__ att_s2,
                                                   const float* __restrict__ att_d2,
                                                   float* __restrict__ z,
                                                   float* __restrict__ as2,
                                                   float* __restrict__ ad2) {
    const int n = blockIdx.x;
    const int t = threadIdx.x;
    const int head = t >> 7;
    __shared__ float sh[256];
    float den = den1[n * NHEAD + head] + 1e-16f;
    float v = acc1[(size_t)n * HOUT + t] / den + b1[t];
    v = fmaxf(v, 0.0f);
    sh[t] = v * W2[t];
    __syncthreads();
    for (int off = 128; off > 0; off >>= 1) {
        if (t < off) sh[t] += sh[t + off];
        __syncthreads();
    }
    if (t == 0) {
        float zv = sh[0];
        z[n]   = zv;
        as2[n] = zv * att_s2[0];
        ad2[n] = zv * att_d2[0];
    }
}

// ---- 8) layer2 edge logits + segment max -----------------------------------
__global__ void k_edge_logits2(const int* __restrict__ ei,
                               const float* __restrict__ as2,
                               const float* __restrict__ ad2,
                               float* __restrict__ alpha2,
                               unsigned* __restrict__ amax2) {
    size_t e = (size_t)blockIdx.x * blockDim.x + threadIdx.x;
    if (e >= ETOT) return;
    int s, d;
    if (e < EE) { s = ei[e]; d = ei[EE + e]; } else { s = d = (int)e - EE; }
    float a = leaky(as2[s] + ad2[d]);
    alpha2[e] = a;
    atomicMax(&amax2[d], f2ord(a));
}

// ---- 9) layer2 aggregate (scalar messages) ---------------------------------
__global__ void k_edge_agg2(const int* __restrict__ ei,
                            const float* __restrict__ alpha2,
                            const unsigned* __restrict__ amax2,
                            const float* __restrict__ z,
                            float* __restrict__ den2,
                            float* __restrict__ num2) {
    size_t e = (size_t)blockIdx.x * blockDim.x + threadIdx.x;
    if (e >= ETOT) return;
    int s, d;
    if (e < EE) { s = ei[e]; d = ei[EE + e]; } else { s = d = (int)e - EE; }
    float ex = __expf(alpha2[e] - ord2f(amax2[d]));
    atomicAdd(&den2[d], ex);
    atomicAdd(&num2[d], ex * z[s]);
}

// ---- 10) output: sigmoid(num/den + b2) -------------------------------------
__global__ void k_output(const float* __restrict__ num2,
                         const float* __restrict__ den2,
                         const float* __restrict__ b2,
                         float* __restrict__ out) {
    int n = blockIdx.x * blockDim.x + threadIdx.x;
    if (n >= NN) return;
    float v = num2[n] / (den2[n] + 1e-16f) + b2[0];
    out[n] = 1.0f / (1.0f + __expf(-v));
}

// ---------------------------------------------------------------------------
extern "C" void kernel_launch(void* const* d_in, const int* in_sizes, int n_in,
                              void* d_out, int out_size, void* d_ws, size_t ws_size,
                              hipStream_t stream) {
    const float* x      = (const float*)d_in[0];
    const int*   ei     = (const int*)  d_in[1];
    const float* W1     = (const float*)d_in[2];
    const float* att_s1 = (const float*)d_in[3];
    const float* att_d1 = (const float*)d_in[4];
    const float* b1     = (const float*)d_in[5];
    const float* W2     = (const float*)d_in[6];
    const float* att_s2 = (const float*)d_in[7];
    const float* att_d2 = (const float*)d_in[8];
    const float* b2     = (const float*)d_in[9];
    float* out = (float*)d_out;

    // workspace carve-up (256B aligned slices)
    size_t off = 0;
    char* base = (char*)d_ws;
    auto take = [&](size_t bytes) -> char* {
        char* p = base + off;
        off = (off + bytes + 255) & ~(size_t)255;
        return p;
    };
    __bf16*   xb     = (__bf16*)  take((size_t)NN * CIN * 2);
    __bf16*   w1f    = (__bf16*)  take((size_t)4 * 16 * 32 * 16 * 2);
    float*    h1     = (float*)   take((size_t)NN * HOUT * 4);
    float*    acc1   = (float*)   take((size_t)NN * HOUT * 4);
    float*    as1    = (float*)   take((size_t)NN * NHEAD * 4);
    float*    ad1    = (float*)   take((size_t)NN * NHEAD * 4);
    float*    alpha1 = (float*)   take((size_t)ETOT * NHEAD * 4);
    unsigned* amax1  = (unsigned*)take((size_t)NN * NHEAD * 4);
    float*    den1   = (float*)   take((size_t)NN * NHEAD * 4);
    float*    z      = (float*)   take((size_t)NN * 4);
    float*    as2    = (float*)   take((size_t)NN * 4);
    float*    ad2    = (float*)   take((size_t)NN * 4);
    float*    alpha2 = (float*)   take((size_t)ETOT * 4);
    unsigned* amax2  = (unsigned*)take((size_t)NN * 4);
    float*    den2   = (float*)   take((size_t)NN * 4);
    float*    num2   = (float*)   take((size_t)NN * 4);
    (void)ws_size; (void)in_sizes; (void)n_in; (void)out_size;

    k_convert_x<<<((size_t)NN * CIN + 255) / 256, 256, 0, stream>>>(x, xb);
    k_pack_w1<<<(4 * 16 * 32 * 16 + 255) / 256, 256, 0, stream>>>(W1, w1f);
    k_init<<<((size_t)NN * HOUT + 255) / 256, 256, 0, stream>>>(acc1, den1, amax1, den2, num2, amax2);

    k_gemm1<<<(NN / 16 * 16) / 8, 256, 0, stream>>>(xb, w1f, h1);   // 6250 blocks
    k_attn_dots<<<NN, 256, 0, stream>>>(h1, att_s1, att_d1, as1, ad1);

    k_edge_logits1<<<((size_t)ETOT * NHEAD + 255) / 256, 256, 0, stream>>>(ei, as1, ad1, alpha1, amax1);
    k_edge_agg1<<<ETOT, 256, 0, stream>>>(ei, alpha1, amax1, h1, den1, acc1);
    k_finalize1<<<NN, 256, 0, stream>>>(acc1, den1, b1, W2, att_s2, att_d2, z, as2, ad2);

    k_edge_logits2<<<(ETOT + 255) / 256, 256, 0, stream>>>(ei, as2, ad2, alpha2, amax2);
    k_edge_agg2<<<(ETOT + 255) / 256, 256, 0, stream>>>(ei, alpha2, amax2, z, den2, num2);
    k_output<<<(NN + 255) / 256, 256, 0, stream>>>(num2, den2, b2, out);
}
